// TritonConv2dINT8_43920335569347
// MI455X (gfx1250) — 8-bit matrix operand order repaired and checked by reading, compiled, not run on hardware
//
#include <hip/hip_runtime.h>
#include <stdint.h>

typedef int v8i __attribute__((ext_vector_type(8)));

#define B_    32
#define CIN_  64
#define COUT_ 64
#define H_    112
#define W_    112
#define HW_   (H_ * W_)
#define XH_BYTES ((size_t)B_ * HW_ * CIN_)
#define WPK_DWORDS (9 * 4 * 32 * 8)

__global__ void pack_weights_kernel(const int8_t* __restrict__ w,
                                    uint32_t* __restrict__ wpk) {
  const int idx = blockIdx.x * blockDim.x + threadIdx.x;
  if (idx >= 9 * 4 * 32) return;
  const int lane = idx & 31;
  const int nt   = (idx >> 5) & 3;
  const int tap  = idx >> 7;
  const int n    = lane & 15;
  const int lh   = lane >> 4;
  const int cout = nt * 16 + n;
  uint32_t* dst = wpk + (size_t)idx * 8;
#pragma unroll
  for (int v = 0; v < 8; ++v) {
    uint32_t d = 0;
#pragma unroll
    for (int bb = 0; bb < 4; ++bb) {
      const int k = 16 * (v >> 1) + 8 * lh + 4 * (v & 1) + bb;
      const uint8_t byte = (uint8_t)w[cout * 576 + k * 9 + tap];
      d |= ((uint32_t)byte) << (8 * bb);
    }
    dst[v] = d;
  }
}

__global__ __launch_bounds__(256) void nchw2nhwc_kernel(
    const int8_t* __restrict__ x, int8_t* __restrict__ xh) {
  __shared__ __align__(16) uint8_t tile[64][80];
  const int b   = blockIdx.x / 196;
  const int hwt = blockIdx.x % 196;
  const int hw0 = hwt * 64;
  const int t = threadIdx.x;
  {
    const int c  = t >> 2;
    const int px = (t & 3) * 16;
    const uint4 v =
        *(const uint4*)(x + ((size_t)(b * CIN_ + c) * HW_ + hw0 + px));
    *(uint4*)&tile[c][px] = v;
  }
  __syncthreads();
  {
    const int px = t >> 2;
    const int c0 = (t & 3) * 16;
    uint32_t d[4];
#pragma unroll
    for (int j = 0; j < 4; ++j) {
      d[j] = (uint32_t)tile[c0 + 4 * j + 0][px] |
             ((uint32_t)tile[c0 + 4 * j + 1][px] << 8) |
             ((uint32_t)tile[c0 + 4 * j + 2][px] << 16) |
             ((uint32_t)tile[c0 + 4 * j + 3][px] << 24);
    }
    uint4 v;
    v.x = d[0]; v.y = d[1]; v.z = d[2]; v.w = d[3];
    *(uint4*)(xh + ((size_t)(b * HW_ + hw0 + px) * CIN_ + c0)) = v;
  }
}

__global__ __launch_bounds__(256) void conv_wmma_kernel(
    const int8_t* __restrict__ xh, const uint32_t* __restrict__ wpk,
    const float* __restrict__ wscale, const float* __restrict__ ascale,
    const float* __restrict__ bias, float* __restrict__ out) {
  __shared__ __align__(16) uint32_t wlds[WPK_DWORDS];
  {
    const uint4* src = (const uint4*)wpk;
    uint4* dst = (uint4*)wlds;
    for (int i = threadIdx.x; i < WPK_DWORDS / 4; i += 256) dst[i] = src[i];
  }
  __syncthreads();

  const int lane = threadIdx.x & 31;
  const int wave = threadIdx.x >> 5;
  const int tilei = blockIdx.x * 8 + wave;
  const int wt = tilei % 7;
  int rem = tilei / 7;
  const int h = rem % H_;
  const int b = rem / H_;
  const int w0 = wt * 16;

  const int row = lane & 15;
  const int lh  = lane >> 4;

  v8i acc0 = {}, acc1 = {}, acc2 = {}, acc3 = {};

#pragma unroll
  for (int tap = 0; tap < 9; ++tap) {
    const int kh = tap / 3, kw = tap % 3;
    const int hin = h + kh - 1;
    const int win = w0 + row + kw - 1;
    const bool valid = ((unsigned)hin < H_) && ((unsigned)win < W_);

    union { unsigned long long q[4]; v8i v; } av;
    if (valid) {
      const int8_t* p =
          xh + ((size_t)(b * HW_ + hin * W_ + win) * CIN_) + lh * 8;
      av.q[0] = *(const unsigned long long*)(p + 0);
      av.q[1] = *(const unsigned long long*)(p + 16);
      av.q[2] = *(const unsigned long long*)(p + 32);
      av.q[3] = *(const unsigned long long*)(p + 48);
    } else {
      av.q[0] = 0; av.q[1] = 0; av.q[2] = 0; av.q[3] = 0;
    }

    const uint32_t* bl = wlds + tap * 1024 + lane * 8;
    union { uint4 u[2]; v8i v; } bv0, bv1, bv2, bv3;
    bv0.u[0] = *(const uint4*)(bl + 0);   bv0.u[1] = *(const uint4*)(bl + 4);
    bv1.u[0] = *(const uint4*)(bl + 256); bv1.u[1] = *(const uint4*)(bl + 260);
    bv2.u[0] = *(const uint4*)(bl + 512); bv2.u[1] = *(const uint4*)(bl + 516);
    bv3.u[0] = *(const uint4*)(bl + 768); bv3.u[1] = *(const uint4*)(bl + 772);

    acc0 = __builtin_amdgcn_wmma_i32_16x16x64_iu8(true, av.v, true, bv0.v,
                                                  acc0, false, false);
    acc1 = __builtin_amdgcn_wmma_i32_16x16x64_iu8(true, av.v, true, bv1.v,
                                                  acc1, false, false);
    acc2 = __builtin_amdgcn_wmma_i32_16x16x64_iu8(true, av.v, true, bv2.v,
                                                  acc2, false, false);
    acc3 = __builtin_amdgcn_wmma_i32_16x16x64_iu8(true, av.v, true, bv3.v,
                                                  acc3, false, false);
  }

  const float s = ascale[0] * wscale[0];
  const int mbase = lh * 8;

#define EPILOGUE(NT, ACC)                                                     \
  {                                                                           \
    const int cout = (NT) * 16 + row;                                         \
    const float bvl = bias[cout];                                             \
    float* o = out + ((size_t)(b * COUT_ + cout) * HW_ + h * W_ + w0 + mbase);\
    float4 lo, hi;                                                            \
    lo.x = (float)ACC[0] * s + bvl;  lo.y = (float)ACC[1] * s + bvl;          \
    lo.z = (float)ACC[2] * s + bvl;  lo.w = (float)ACC[3] * s + bvl;          \
    hi.x = (float)ACC[4] * s + bvl;  hi.y = (float)ACC[5] * s + bvl;          \
    hi.z = (float)ACC[6] * s + bvl;  hi.w = (float)ACC[7] * s + bvl;          \
    *(float4*)(o + 0) = lo;                                                   \
    *(float4*)(o + 4) = hi;                                                   \
  }

  EPILOGUE(0, acc0)
  EPILOGUE(1, acc1)
  EPILOGUE(2, acc2)
  EPILOGUE(3, acc3)
#undef EPILOGUE
}

extern "C" void kernel_launch(void* const* d_in, const int* in_sizes, int n_in,
                              void* d_out, int out_size, void* d_ws,
                              size_t ws_size, hipStream_t stream) {
  const int8_t* x_q    = (const int8_t*)d_in[0];
  const int8_t* w_q    = (const int8_t*)d_in[1];
  const float*  wscale = (const float*)d_in[2];
  const float*  ascale = (const float*)d_in[3];
  const float*  bias   = (const float*)d_in[4];
  float* out = (float*)d_out;

  int8_t*   xh  = (int8_t*)d_ws;
  uint32_t* wpk = (uint32_t*)((char*)d_ws + XH_BYTES);

  pack_weights_kernel<<<9, 128, 0, stream>>>(w_q, wpk);
  nchw2nhwc_kernel<<<B_ * 196, 256, 0, stream>>>(x_q, xh);
  conv_wmma_kernel<<<(B_ * H_ * 7) / 8, 256, 0, stream>>>(xh, wpk, wscale,
                                                          ascale, bias, out);
}
